// LSTMModel_12369505813017
// MI455X (gfx1250) — matrix operand order repaired and checked by reading, not compiled, not run on hardware
//
#include <hip/hip_runtime.h>

typedef __attribute__((ext_vector_type(16))) __bf16 v16bf;
typedef __attribute__((ext_vector_type(8)))  __bf16 v8bf;
typedef __attribute__((ext_vector_type(8)))  float  v8f;

// ---------- helpers ----------

__device__ __forceinline__ __bf16 f2bf(float f) {
  union { float f; unsigned u; } v; v.f = f;
  unsigned r = v.u + 0x7FFFu + ((v.u >> 16) & 1u);   // round-to-nearest-even
  unsigned short h = (unsigned short)(r >> 16);
  __bf16 out;
  __builtin_memcpy(&out, &h, 2);
  return out;
}

__device__ __forceinline__ float sigm(float x) {
  return 1.0f / (1.0f + __expf(-x));
}
__device__ __forceinline__ float tanh_fast(float x) {
  float xc = fminf(fmaxf(x, -15.0f), 15.0f);
  float e = __expf(2.0f * xc);
  return (e - 1.0f) / (e + 1.0f);
}

// A fragment: 16x32 bf16 tile at (row0,k0) of row-major A[*, lda].
// ISA layout: lanes 0-15 -> M=lane, K = k0+{0..7, 16..23};
//             lanes 16-31 -> M=lane-16, K = k0+{8..15, 24..31}.
__device__ __forceinline__ v16bf load_frag_a(const __bf16* __restrict__ A,
                                             int lda, int row0, int k0, int lane) {
  int r  = row0 + (lane & 15);
  int kb = k0 + ((lane & 16) ? 8 : 0);
  const v8bf* p = (const v8bf*)(A + (size_t)r * lda + kb);
  v8bf lo = p[0];        // K = kb .. kb+7      (16 contiguous bytes)
  v8bf hi = p[2];        // K = kb+16 .. kb+23  (16 contiguous bytes)
  return __builtin_shufflevector(lo, hi, 0,1,2,3,4,5,6,7,8,9,10,11,12,13,14,15);
}

// B fragment: 32x16 bf16 tile; B stored pre-transposed as Bt[N, ldk] (row = N col).
// ISA layout: lanes 0-15 -> N=lane, K = k0+{0..7, 16..23}; lanes 16-31 -> N=lane-16, K = k0+{8..15, 24..31}.
__device__ __forceinline__ v16bf load_frag_b(const __bf16* __restrict__ Bt,
                                             int ldk, int n0, int k0, int lane) {
  int n  = n0 + (lane & 15);
  int kb = k0 + ((lane & 16) ? 8 : 0);
  const v8bf* p = (const v8bf*)(Bt + (size_t)n * ldk + kb);
  v8bf lo = p[0];
  v8bf hi = p[2];
  return __builtin_shufflevector(lo, hi, 0,1,2,3,4,5,6,7,8,9,10,11,12,13,14,15);
}

__device__ __forceinline__ v8f wmma_bf16(v16bf a, v16bf b, v8f c) {
  return __builtin_amdgcn_wmma_f32_16x16x32_bf16(
      /*neg_a=*/false, a, /*neg_b=*/false, b,
      /*c_mod=*/(short)0, c, /*reuse_a=*/false, /*reuse_b=*/false);
}

// ---------- conversion / transpose kernels ----------

__global__ void convert_f32_bf16(const float* __restrict__ src,
                                 __bf16* __restrict__ dst, size_t n) {
  size_t i = (size_t)blockIdx.x * blockDim.x + threadIdx.x;
  size_t stride = (size_t)gridDim.x * blockDim.x;
  for (; i < n; i += stride) dst[i] = f2bf(src[i]);
}

// W[K,N] fp32 row-major -> Wt[N,K] bf16 row-major. K,N multiples of 32.
__global__ __launch_bounds__(256) void transpose_f32_to_bf16(
    const float* __restrict__ W, __bf16* __restrict__ Wt, int K, int N) {
  __shared__ float tile[32][33];
  int nx = blockIdx.x * 32 + threadIdx.x;   // column (N)
  int ky = blockIdx.y * 32 + threadIdx.y;   // row (K)
  for (int j = 0; j < 32; j += 8)
    tile[threadIdx.y + j][threadIdx.x] = W[(size_t)(ky + j) * N + nx];
  __syncthreads();
  int kx = blockIdx.y * 32 + threadIdx.x;   // K
  int ny = blockIdx.x * 32 + threadIdx.y;   // N
  for (int j = 0; j < 32; j += 8)
    Wt[(size_t)(ny + j) * K + kx] = f2bf(tile[threadIdx.x][threadIdx.y + j]);
}

__global__ void init_state(__bf16* __restrict__ h, float* __restrict__ c, int n) {
  int i = blockIdx.x * blockDim.x + threadIdx.x;
  if (i < n) { h[i] = f2bf(0.0f); c[i] = 0.0f; }
}

// ---------- big GEMM: C[M,N] = A[M,K] * Bt[N,K]^T + bias[N] ----------
// block: 256 threads = 8 waves; wave -> 32x32 fp32 tile; block -> 64x128.
__global__ __launch_bounds__(256) void gemm_bias_wmma(
    const __bf16* __restrict__ A, const __bf16* __restrict__ Bt,
    const float* __restrict__ bias, float* __restrict__ C,
    int M, int N, int K) {
  int tid  = threadIdx.x;
  int wave = tid >> 5;
  int lane = tid & 31;
  int m0 = blockIdx.y * 64  + (wave & 1) * 32;
  int n0 = blockIdx.x * 128 + (wave >> 1) * 32;

  v8f acc00 = {}, acc01 = {}, acc10 = {}, acc11 = {};
  for (int k = 0; k < K; k += 32) {
    __builtin_prefetch(A  + (size_t)(m0 + (lane & 15)) * K + k + 32, 0, 1);
    __builtin_prefetch(Bt + (size_t)(n0 + (lane & 15)) * K + k + 32, 0, 1);
    v16bf a0 = load_frag_a(A, K, m0,      k, lane);
    v16bf a1 = load_frag_a(A, K, m0 + 16, k, lane);
    v16bf b0 = load_frag_b(Bt, K, n0,      k, lane);
    v16bf b1 = load_frag_b(Bt, K, n0 + 16, k, lane);
    acc00 = wmma_bf16(a0, b0, acc00);
    acc01 = wmma_bf16(a0, b1, acc01);
    acc10 = wmma_bf16(a1, b0, acc10);
    acc11 = wmma_bf16(a1, b1, acc11);
  }

  int col = lane & 15;
  int rb  = (lane & 16) ? 8 : 0;
  float bb0 = bias[n0 + col];
  float bb1 = bias[n0 + 16 + col];
#pragma unroll
  for (int v = 0; v < 8; ++v) {
    C[(size_t)(m0 + rb + v)      * N + n0 + col]      = acc00[v] + bb0;
    C[(size_t)(m0 + rb + v)      * N + n0 + 16 + col] = acc01[v] + bb1;
    C[(size_t)(m0 + 16 + rb + v) * N + n0 + col]      = acc10[v] + bb0;
    C[(size_t)(m0 + 16 + rb + v) * N + n0 + 16 + col] = acc11[v] + bb1;
  }
}

// ---------- LSTM recurrent step ----------
// grid = F/16 blocks; block = 8 waves: wave (gate = w>>1, mhalf = w&1) computes the
// 16x16 z-tile of its gate for its 16 feature columns; fused gate math after barrier.
__global__ __launch_bounds__(256) void lstm_step_wmma(
    const float*  __restrict__ xz,    // [B*T, 4F], row m = b*T + t
    const __bf16* __restrict__ Wht,   // [4F, F] (transposed Wh)
    const __bf16* __restrict__ h_in,  // [B, F]
    __bf16*       __restrict__ h_out, // [B, F]
    float*        __restrict__ c,     // [B, F]
    float*        __restrict__ y,     // [B*T, F] fp32 or nullptr
    __bf16*       __restrict__ ybf,   // [B*T, F] bf16
    int t) {
  constexpr int Bsz = 32, T = 512, F = 1024;
  __shared__ float zs[4][32][16];

  int tid  = threadIdx.x;
  int wave = tid >> 5;
  int lane = tid & 31;
  int gate = wave >> 1;
  int mt   = (wave & 1) * 16;
  int j0   = blockIdx.x * 16;
  int n0   = gate * F + j0;

  v8f acc = {};
  for (int k = 0; k < F; k += 32) {
    v16bf a = load_frag_a(h_in, F, mt, k, lane);
    v16bf b = load_frag_b(Wht, F, n0, k, lane);
    acc = wmma_bf16(a, b, acc);
  }

  int col = lane & 15;
  int rb  = (lane & 16) ? 8 : 0;
#pragma unroll
  for (int v = 0; v < 8; ++v) {
    int bidx = mt + rb + v;                              // batch row 0..31
    float z = acc[v] + xz[((size_t)bidx * T + t) * (4 * F) + n0 + col];
    zs[gate][bidx][col] = z;
  }
  __syncthreads();

  for (int e = tid; e < Bsz * 16; e += 256) {
    int b  = e >> 4;
    int jj = e & 15;
    int j  = j0 + jj;
    float zi = zs[0][b][jj], zf = zs[1][b][jj];
    float zg = zs[2][b][jj], zo = zs[3][b][jj];
    float cprev = c[b * F + j];
    float cn = sigm(zf) * cprev + sigm(zi) * tanh_fast(zg);
    float hn = sigm(zo) * tanh_fast(cn);
    c[b * F + j] = cn;
    __bf16 hb = f2bf(hn);
    h_out[b * F + j] = hb;
    size_t yi = ((size_t)b * T + t) * F + j;
    ybf[yi] = hb;
    if (y) y[yi] = hn;
  }
}

// ---------- host launch ----------

extern "C" void kernel_launch(void* const* d_in, const int* in_sizes, int n_in,
                              void* d_out, int out_size, void* d_ws, size_t ws_size,
                              hipStream_t stream) {
  (void)in_sizes; (void)n_in; (void)out_size; (void)ws_size;
  constexpr int    Bsz = 32, T = 512, F = 1024, G = 4 * F;
  constexpr size_t MT  = (size_t)Bsz * T;

  const float* x   = (const float*)d_in[0];
  const float* Wi0 = (const float*)d_in[1];
  const float* Wh0 = (const float*)d_in[2];
  const float* b0  = (const float*)d_in[3];
  const float* Wi1 = (const float*)d_in[4];
  const float* Wh1 = (const float*)d_in[5];
  const float* b1  = (const float*)d_in[6];
  float* out = (float*)d_out;

  char* ws = (char*)d_ws;
  float*  xz   = (float*) ws;  ws += MT * G * sizeof(float);   // 256 MB
  __bf16* xbf  = (__bf16*)ws;  ws += MT * F * sizeof(__bf16);  //  32 MB
  __bf16* ybf  = (__bf16*)ws;  ws += MT * F * sizeof(__bf16);  //  32 MB
  __bf16* Wi0t = (__bf16*)ws;  ws += (size_t)G * F * 2;        //   8 MB each
  __bf16* Wh0t = (__bf16*)ws;  ws += (size_t)G * F * 2;
  __bf16* Wi1t = (__bf16*)ws;  ws += (size_t)G * F * 2;
  __bf16* Wh1t = (__bf16*)ws;  ws += (size_t)G * F * 2;
  __bf16* h0   = (__bf16*)ws;  ws += (size_t)Bsz * F * 2;
  __bf16* h1   = (__bf16*)ws;  ws += (size_t)Bsz * F * 2;
  float*  cbuf = (float*) ws;  ws += (size_t)Bsz * F * sizeof(float);

  // 1) convert inputs / weights to bf16 (weights pre-transposed to [N,K])
  convert_f32_bf16<<<4096, 256, 0, stream>>>(x, xbf, MT * F);
  dim3 tb(32, 8), tg(G / 32, F / 32);
  transpose_f32_to_bf16<<<tg, tb, 0, stream>>>(Wi0, Wi0t, F, G);
  transpose_f32_to_bf16<<<tg, tb, 0, stream>>>(Wh0, Wh0t, F, G);
  transpose_f32_to_bf16<<<tg, tb, 0, stream>>>(Wi1, Wi1t, F, G);
  transpose_f32_to_bf16<<<tg, tb, 0, stream>>>(Wh1, Wh1t, F, G);

  dim3 gg(G / 128, (int)(MT / 64));   // (32, 256) blocks
  int  nstate = Bsz * F;

  // ---- layer 0 ----
  gemm_bias_wmma<<<gg, 256, 0, stream>>>(xbf, Wi0t, b0, xz, (int)MT, G, F);
  init_state<<<(nstate + 255) / 256, 256, 0, stream>>>(h0, cbuf, nstate);
  for (int t = 0; t < T; ++t) {
    const __bf16* hi = (t & 1) ? h1 : h0;
    __bf16*       ho = (t & 1) ? h0 : h1;
    lstm_step_wmma<<<F / 16, 256, 0, stream>>>(xz, Wh0t, hi, ho, cbuf,
                                               nullptr, ybf, t);
  }

  // ---- layer 1 ----
  gemm_bias_wmma<<<gg, 256, 0, stream>>>(ybf, Wi1t, b1, xz, (int)MT, G, F);
  init_state<<<(nstate + 255) / 256, 256, 0, stream>>>(h0, cbuf, nstate);
  for (int t = 0; t < T; ++t) {
    const __bf16* hi = (t & 1) ? h1 : h0;
    __bf16*       ho = (t & 1) ? h0 : h1;
    lstm_step_wmma<<<F / 16, 256, 0, stream>>>(xz, Wh1t, hi, ho, cbuf,
                                               out, ybf, t);
  }
}
